// GCNlstm_static_49340584296687
// MI455X (gfx1250) — matrix operand order repaired and checked by reading, not compiled, not run on hardware
//
#include <hip/hip_runtime.h>
#include <cstdint>
#include <cstddef>

#define NNODES 80000
#define NF     128      // NFEAT == D1 == D2 == 128
#define GATES  512      // 4*D2
#define NMETA  4
#define HOUSE  5000
#define SEQLEN (NNODES / HOUSE)   // 16

typedef __attribute__((ext_vector_type(16))) __bf16 v16bf;
typedef __attribute__((ext_vector_type(8)))  float  v8f;

// ---------------------------------------------------------------------------
// WMMA GEMM:  C[M,N] (f32, ldc=N) = A[M,K] (f32) @ B[K,N], where B is supplied
// pre-transposed+converted as Bt[N,K] in bf16.  K multiple of 32, N mult of 64.
// One wave -> 16x64 C tile (4 WMMAs per K-step sharing one A fragment);
// block = 4 waves stacked along M -> 64x64 block tile.
// ---------------------------------------------------------------------------
__global__ __launch_bounds__(128)
void wmma_gemm_bf16(const float* __restrict__ A, const __bf16* __restrict__ Bt,
                    float* __restrict__ C, int M, int N, int K) {
  const int m0 = (blockIdx.x * 4 + threadIdx.y) * 16;   // wave's 16 rows
  const int n0 = blockIdx.y * 64;                       // wave's 64 cols
  const int lane = threadIdx.x;                         // 0..31
  const int l    = lane & 15;
  const int hi   = lane >> 4;                           // 0 or 1

  int arow = m0 + l;
  if (arow >= M) arow = M - 1;                          // clamp (stores guarded)
  const float*  ap = A  + (size_t)arow * K + hi * 16;
  const __bf16* bp = Bt + (size_t)(n0 + l) * K + hi * 16;
  const size_t  bstep = (size_t)16 * K;                 // next 16-column group

  v8f acc0 = {}, acc1 = {}, acc2 = {}, acc3 = {};
  #pragma unroll 2
  for (int kk = 0; kk < K; kk += 32) {
    // A fragment (16x32 bf16): lane<16 -> K kk+0..15 of row m0+l
    //                          lane>=16 -> K kk+16..31
    float4 a0 = *(const float4*)(ap + kk);
    float4 a1 = *(const float4*)(ap + kk + 4);
    float4 a2 = *(const float4*)(ap + kk + 8);
    float4 a3 = *(const float4*)(ap + kk + 12);
    v16bf av;
    av[0]=(__bf16)a0.x;  av[1]=(__bf16)a0.y;  av[2]=(__bf16)a0.z;  av[3]=(__bf16)a0.w;
    av[4]=(__bf16)a1.x;  av[5]=(__bf16)a1.y;  av[6]=(__bf16)a1.z;  av[7]=(__bf16)a1.w;
    av[8]=(__bf16)a2.x;  av[9]=(__bf16)a2.y;  av[10]=(__bf16)a2.z; av[11]=(__bf16)a2.w;
    av[12]=(__bf16)a3.x; av[13]=(__bf16)a3.y; av[14]=(__bf16)a3.z; av[15]=(__bf16)a3.w;

    // B fragments (32x16 bf16 each): lane<16 -> K kk+0..15 of col; >=16 -> kk+16..31
    v16bf b0 = *(const v16bf*)(bp             + kk);
    v16bf b1 = *(const v16bf*)(bp + 1 * bstep + kk);
    v16bf b2 = *(const v16bf*)(bp + 2 * bstep + kk);
    v16bf b3 = *(const v16bf*)(bp + 3 * bstep + kk);

    acc0 = __builtin_amdgcn_wmma_f32_16x16x32_bf16(false, av, false, b0, (short)0, acc0, false, false);
    acc1 = __builtin_amdgcn_wmma_f32_16x16x32_bf16(false, av, false, b1, (short)0, acc1, false, false);
    acc2 = __builtin_amdgcn_wmma_f32_16x16x32_bf16(false, av, false, b2, (short)0, acc2, false, false);
    acc3 = __builtin_amdgcn_wmma_f32_16x16x32_bf16(false, av, false, b3, (short)0, acc3, false, false);
  }

  // C layout: VGPR v -> row m0 + v + hi*8, col n0 + l (+16 per tile)
  const int crow = m0 + hi * 8;
  const int ccol = n0 + l;
  #pragma unroll
  for (int v = 0; v < 8; ++v) {
    int r = crow + v;
    if (r < M) {
      float* crp = C + (size_t)r * N + ccol;
      crp[0]  = acc0[v];
      crp[16] = acc1[v];
      crp[32] = acc2[v];
      crp[48] = acc3[v];
    }
  }
}

// ---------------------------------------------------------------------------
// Edge-parallel SpMM: out[rows[e], :] += vals[e] * mat[cols[e], :]
// One wave per edge, lane handles 4 consecutive features (128 total).
// Gather/scatter tables (41 MB) live in the 192 MB L2; atomics resolve there.
// ---------------------------------------------------------------------------
__global__ __launch_bounds__(256)
void spmm_edges(const int* __restrict__ rows, const int* __restrict__ cols,
                const float* __restrict__ vals, const float* __restrict__ mat,
                float* __restrict__ out, int nedges) {
  int edge = blockIdx.x * 8 + (threadIdx.x >> 5);
  if (edge >= nedges) return;
  int lane = threadIdx.x & 31;

  int   r = rows[edge];
  int   c = cols[edge];
  float v = vals[edge];

  // Hint a future gather row toward the caches (global_prefetch_b8 path).
  int pe = edge + 4096;
  if (pe < nedges && lane == 0)
    __builtin_prefetch(mat + (size_t)cols[pe] * NF, 0, 1);

  const float4 d = ((const float4*)(mat + (size_t)c * NF))[lane];
  float* dst = out + (size_t)r * NF + lane * 4;
  atomicAdd(dst + 0, v * d.x);
  atomicAdd(dst + 1, v * d.y);
  atomicAdd(dst + 2, v * d.z);
  atomicAdd(dst + 3, v * d.w);
}

// ---------------------------------------------------------------------------
// Small elementwise / prep kernels
// ---------------------------------------------------------------------------
__global__ void zero_f32(float* __restrict__ p, size_t n) {
  size_t i = (size_t)blockIdx.x * blockDim.x + threadIdx.x;
  size_t stride = (size_t)gridDim.x * blockDim.x;
  for (; i < n; i += stride) p[i] = 0.0f;
}

// dst[n*K + k] = (bf16) src[k*N + n]
__global__ void transpose_to_bf16(const float* __restrict__ src, __bf16* __restrict__ dst,
                                  int K, int N) {
  int i = blockIdx.x * blockDim.x + threadIdx.x;
  if (i >= K * N) return;
  int n = i / K, k = i - n * K;
  dst[i] = (__bf16)src[(size_t)k * N + n];
}

__global__ void convert_to_bf16(const float* __restrict__ src, __bf16* __restrict__ dst, int n) {
  int i = blockIdx.x * blockDim.x + threadIdx.x;
  if (i < n) dst[i] = (__bf16)src[i];
}

// h = relu(h + b[col])
__global__ void bias_relu_inplace(float* __restrict__ h, const float* __restrict__ b, size_t total) {
  size_t i = (size_t)blockIdx.x * blockDim.x + threadIdx.x;
  size_t stride = (size_t)gridDim.x * blockDim.x;
  for (; i < total; i += stride)
    h[i] = fmaxf(h[i] + b[i & (NF - 1)], 0.0f);
}

// mix += W_meta[m] * relu(acc + b2[col])
__global__ void mix_accum(float* __restrict__ mix, const float* __restrict__ acc,
                          const float* __restrict__ b2, const float* __restrict__ wmeta,
                          int m, size_t total) {
  float wm = wmeta[m];
  size_t i = (size_t)blockIdx.x * blockDim.x + threadIdx.x;
  size_t stride = (size_t)gridDim.x * blockDim.x;
  for (; i < total; i += stride)
    mix[i] += wm * fmaxf(acc[i] + b2[i & (NF - 1)], 0.0f);
}

__global__ void relu_inplace(float* __restrict__ p, size_t n) {
  size_t i = (size_t)blockIdx.x * blockDim.x + threadIdx.x;
  size_t stride = (size_t)gridDim.x * blockDim.x;
  for (; i < n; i += stride) p[i] = fmaxf(p[i], 0.0f);
}

// ---------------------------------------------------------------------------
// Fused LSTM cell + output projection.  One block (128 threads) per batch row.
// gx = mixed_t @ Wih^T, gh = h @ Whh^T, gate = gx + gh + bih + bhh.
// Also emits out[t*HOUSE + b] = dot(h_new, lin_w) + lin_b via LDS reduction.
// ---------------------------------------------------------------------------
__device__ __forceinline__ float sigmoidf(float x) { return 1.0f / (1.0f + __expf(-x)); }

__global__ __launch_bounds__(NF)
void lstm_cell(const float* __restrict__ gx, const float* __restrict__ gh,
               const float* __restrict__ bih, const float* __restrict__ bhh,
               float* __restrict__ h, float* __restrict__ c,
               const float* __restrict__ lin_w, const float* __restrict__ lin_b,
               float* __restrict__ out, int t) {
  const int b = blockIdx.x;       // 0..HOUSE-1
  const int j = threadIdx.x;      // 0..127 (feature)
  const float* gxr = gx + (size_t)b * GATES;
  const float* ghr = gh + (size_t)b * GATES;

  float gi = sigmoidf(gxr[j          ] + ghr[j          ] + bih[j          ] + bhh[j          ]);
  float gf = sigmoidf(gxr[j +     NF ] + ghr[j +     NF ] + bih[j +     NF ] + bhh[j +     NF ]);
  float gg = tanhf   (gxr[j + 2 * NF ] + ghr[j + 2 * NF ] + bih[j + 2 * NF ] + bhh[j + 2 * NF ]);
  float go = sigmoidf(gxr[j + 3 * NF ] + ghr[j + 3 * NF ] + bih[j + 3 * NF ] + bhh[j + 3 * NF ]);

  size_t idx = (size_t)b * NF + j;
  float cn = gf * c[idx] + gi * gg;
  float hn = go * tanhf(cn);
  c[idx] = cn;
  h[idx] = hn;

  __shared__ float red[NF];
  red[j] = hn * lin_w[j];
  __syncthreads();
  #pragma unroll
  for (int s = NF / 2; s > 0; s >>= 1) {
    if (j < s) red[j] += red[j + s];
    __syncthreads();
  }
  if (j == 0) out[(size_t)t * HOUSE + b] = red[0] + lin_b[0];
}

// ---------------------------------------------------------------------------
// Host orchestration
// ---------------------------------------------------------------------------
extern "C" void kernel_launch(void* const* d_in, const int* in_sizes, int n_in,
                              void* d_out, int out_size, void* d_ws, size_t ws_size,
                              hipStream_t stream) {
  const float* x         = (const float*)d_in[0];
  const int*   edge_rows = (const int*)  d_in[1];
  const int*   edge_cols = (const int*)  d_in[2];
  const float* edge_vals = (const float*)d_in[3];
  const float* gc1_w     = (const float*)d_in[4];
  const float* gc1_b     = (const float*)d_in[5];
  const float* gc2_w     = (const float*)d_in[6];
  const float* gc2_b     = (const float*)d_in[7];
  const float* W_meta    = (const float*)d_in[8];
  const float* lstm_Wih  = (const float*)d_in[9];
  const float* lstm_Whh  = (const float*)d_in[10];
  const float* lstm_bih  = (const float*)d_in[11];
  const float* lstm_bhh  = (const float*)d_in[12];
  const float* lin_w     = (const float*)d_in[13];
  const float* lin_b     = (const float*)d_in[14];

  const int E = in_sizes[1] / NMETA;               // edges per meta-graph
  const size_t NB = (size_t)NNODES * NF;           // 10,240,000 floats per [N,128]

  // -------- workspace layout (floats) --------
  float* f   = (float*)d_ws;
  float* XW  = f;                 // [N,128]  x @ gc1_w
  float* T1  = f + 1 * NB;        // [N,128]  spmm accumulator / h1 / h2-acc
  float* T2  = f + 2 * NB;        // [N,128]  h1 @ gc2_w
  float* MIX = f + 3 * NB;        // [N,128]  meta-mixed features
  float* GX  = f + 4 * NB;                            // [5000,512]
  float* GH  = GX + (size_t)HOUSE * GATES;            // [5000,512]
  float* HS  = GH + (size_t)HOUSE * GATES;            // [5000,128]
  float* CS  = HS + (size_t)HOUSE * NF;               // [5000,128]
  __bf16* bfbase = (__bf16*)(CS + (size_t)HOUSE * NF);
  __bf16* gc1bt  = bfbase;                 // [128,128] = gc1_w^T
  __bf16* gc2bt  = gc1bt + NF * NF;        // [128,128] = gc2_w^T
  __bf16* wihbt  = gc2bt + NF * NF;        // [512,128] = Wih (== (Wih^T)^T)
  __bf16* whhbt  = wihbt + GATES * NF;     // [512,128] = Whh

  const dim3 gblk(32, 4, 1);
  const int ewGrid = 2048;       // grid-stride elementwise grid

  // -------- weight prep (f32 -> bf16, [N,K] layout for GEMM B operand) ------
  transpose_to_bf16<<<(NF * NF + 255) / 256, 256, 0, stream>>>(gc1_w, gc1bt, NF, NF);
  transpose_to_bf16<<<(NF * NF + 255) / 256, 256, 0, stream>>>(gc2_w, gc2bt, NF, NF);
  convert_to_bf16  <<<(GATES * NF + 255) / 256, 256, 0, stream>>>(lstm_Wih, wihbt, GATES * NF);
  convert_to_bf16  <<<(GATES * NF + 255) / 256, 256, 0, stream>>>(lstm_Whh, whhbt, GATES * NF);

  // -------- XW = x @ gc1_w (shared across metas) ----------------------------
  {
    dim3 grid((NNODES + 63) / 64, NF / 64);
    wmma_gemm_bf16<<<grid, gblk, 0, stream>>>(x, gc1bt, XW, NNODES, NF, NF);
  }

  zero_f32<<<ewGrid, 256, 0, stream>>>(MIX, NB);

  // -------- per-meta GCN layers ---------------------------------------------
  for (int m = 0; m < NMETA; ++m) {
    const int*   rs = edge_rows + (size_t)m * E;
    const int*   cs = edge_cols + (size_t)m * E;
    const float* vs = edge_vals + (size_t)m * E;
    dim3 sgrid((E + 7) / 8);
    dim3 ggrid((NNODES + 63) / 64, NF / 64);

    zero_f32<<<ewGrid, 256, 0, stream>>>(T1, NB);
    spmm_edges<<<sgrid, 256, 0, stream>>>(rs, cs, vs, XW, T1, E);
    bias_relu_inplace<<<ewGrid, 256, 0, stream>>>(T1, gc1_b, NB);        // h1
    wmma_gemm_bf16<<<ggrid, gblk, 0, stream>>>(T1, gc2bt, T2, NNODES, NF, NF);
    zero_f32<<<ewGrid, 256, 0, stream>>>(T1, NB);
    spmm_edges<<<sgrid, 256, 0, stream>>>(rs, cs, vs, T2, T1, E);
    mix_accum<<<ewGrid, 256, 0, stream>>>(MIX, T1, gc2_b, W_meta, m, NB);
  }
  relu_inplace<<<ewGrid, 256, 0, stream>>>(MIX, NB);

  // -------- LSTM over SEQLEN steps, batch HOUSE, hidden 128 -----------------
  zero_f32<<<ewGrid, 256, 0, stream>>>(HS, 2 * (size_t)HOUSE * NF);  // h and c (contiguous)

  dim3 lgrid((HOUSE + 63) / 64, GATES / 64);
  for (int t = 0; t < SEQLEN; ++t) {
    const float* xt = MIX + (size_t)t * HOUSE * NF;                  // [5000,128]
    wmma_gemm_bf16<<<lgrid, gblk, 0, stream>>>(xt, wihbt, GX, HOUSE, GATES, NF);
    wmma_gemm_bf16<<<lgrid, gblk, 0, stream>>>(HS, whhbt, GH, HOUSE, GATES, NF);
    lstm_cell<<<HOUSE, NF, 0, stream>>>(GX, GH, lstm_bih, lstm_bhh, HS, CS,
                                        lin_w, lin_b, (float*)d_out, t);
  }
}
